// Feature_Extractor_75694503625223
// MI455X (gfx1250) — matrix operand order repaired and checked by reading, not compiled, not run on hardware
//
#include <hip/hip_runtime.h>
#include <hip/hip_bf16.h>
#include <math.h>
#include <stdint.h>

// ---------------------------------------------------------------------------
// MI455X / gfx1250 point-cloud feature extractor.
// Dense layers use a WMMA GEMM (v_wmma_f32_16x16x32_f16, fp32 accumulate):
//  - activations/weights pre-converted to f16, zero-padded to K%32==0
//  - A/W tiles staged into LDS by the Tensor Data Mover (tensor_load_to_lds),
//    double-buffered, synchronized with s_wait_tensorcnt
//  - per wave: 2 accumulators (32x16), block tile 128x32, guard-free hot loop
// Irregular stages (FPS, kNN, 3x3 eigvals, softmax attn, layernorm, pooling)
// are wave32 VALU/LDS kernels.
// ---------------------------------------------------------------------------

typedef __attribute__((ext_vector_type(16))) _Float16 v16h;
typedef __attribute__((ext_vector_type(8)))  float    v8f;
typedef __attribute__((ext_vector_type(4)))  unsigned int v4u;
typedef __attribute__((ext_vector_type(8)))  int      v8i;
typedef __attribute__((ext_vector_type(4)))  int      v4i;

#define BN_INV_C 0.99999500003750f   // 1/sqrt(1+1e-5)

// ---- Tensor Data Mover: 2D tile (tile_d0 x tile_d1 elems of 2B) -> LDS -----
// D# layout per CDNA5 ISA. pad_enable inserts 1 DWORD
// after every 16 DWORDs so a 32-half row lands with a 34-half LDS pitch.
// clang-23 toolchain: 6-arg builtin (g0, g1, g2, g3, g_extra, cpol).
__device__ static inline void tdm_load_2d(const _Float16* gaddr, unsigned lds_off,
                                          unsigned tensor_d0, unsigned tensor_d1,
                                          unsigned tile_d0, unsigned tile_d1) {
  unsigned long long ga = (unsigned long long)(size_t)gaddr;
  v4u g0;
  g0[0] = 1u;                                         // count=1 (valid user D#)
  g0[1] = lds_off;                                    // lds_addr (bytes)
  g0[2] = (unsigned)(ga & 0xffffffffu);               // global_addr[31:0]
  g0[3] = (unsigned)((ga >> 32) & 0x01ffffffu)        // global_addr[56:32]
        | 0x80000000u;                                // type=2 ("image")
  v8i g1;
  g1[0] = (int)((1u << 16) | (1u << 20) | (3u << 22));// data_size=2B, pad_en,
                                                      // pad_interval=16dw, +1dw
  g1[1] = (int)((tensor_d0 & 0xffffu) << 16);         // tensor_dim0[15:0]
  g1[2] = (int)(((tensor_d0 >> 16) & 0xffffu) | ((tensor_d1 & 0xffffu) << 16));
  g1[3] = (int)(((tensor_d1 >> 16) & 0xffffu) | ((tile_d0 & 0xffffu) << 16));
  g1[4] = (int)(tile_d1 & 0xffffu);                   // tile_dim1; tile_dim2=0
  g1[5] = (int)tensor_d0;                             // dim0 stride (elems)
  g1[6] = 0;
  g1[7] = 0;
  v4i z4 = {0, 0, 0, 0};
  v8i z8 = {0, 0, 0, 0, 0, 0, 0, 0};
  __builtin_amdgcn_tensor_load_to_lds(g0, g1, z4, z4, z8, 0);
}

// f32 (rows,K) -> f16 (rows,K32) zero-padded
__global__ void k_cvt_pad(const float* __restrict__ src, _Float16* __restrict__ dst,
                          int K, int K32, long total) {
  long i = (long)blockIdx.x * blockDim.x + threadIdx.x;
  if (i >= total) return;
  int kk = (int)(i % K32);
  long r = i / K32;
  dst[i] = (kk < K) ? (_Float16)src[r * K + kk] : (_Float16)0.f;
}

// ============================ WMMA GEMM ====================================
// C[m,n] = epilogue( sum_k A[m,k] * W[n,k] + bias[n] )   (A,W already f16)
// act: 0=none 1=relu(bn_eval) 2=relu 3=gelu. resid added after activation.
// Requires: Mrows%128==0, N%32==0, K32%32==0 (guaranteed by the pipeline).
__global__ void __launch_bounds__(256)
k_gemm_wmma_f16(const _Float16* __restrict__ A, const _Float16* __restrict__ W,
                const float* __restrict__ bias, const float* __restrict__ bng,
                const float* __restrict__ bnb, const float* __restrict__ resid,
                float* __restrict__ C, int Mrows, int N, int K32, int act) {
  __shared__ _Float16 As[2][128][34];
  __shared__ _Float16 Ws[2][32][34];

  const int tid  = threadIdx.x;
  const int lane = tid & 31;
  const int wv   = tid >> 5;       // 8 waves
  const int wm   = wv & 3;         // wave row-tile (4 x 32 rows)
  const int wn   = wv >> 2;        // wave col-tile (2 x 16 cols)
  const int blockRow = blockIdx.x * 128;
  const int tileN    = blockIdx.y * 32;

  const int ncol = lane & 15;
  const int akb  = (lane & 16) ? 8 : 0;    // A-frag K-half select
  const int bkb  = (lane & 16) ? 16 : 0;   // unused (B-frag uses ka, same as A)
  const int arow = (wm << 5) + ncol;       // A row within block tile

  v8f acc[2];
#pragma unroll
  for (int v = 0; v < 8; ++v) { acc[0][v] = 0.f; acc[1][v] = 0.f; }

  const int nk = K32 >> 5;
  if (wv == 0) {
    tdm_load_2d(A + (size_t)blockRow * K32, (unsigned)(size_t)&As[0][0][0],
                (unsigned)K32, (unsigned)Mrows, 32, 128);
    tdm_load_2d(W + (size_t)tileN * K32, (unsigned)(size_t)&Ws[0][0][0],
                (unsigned)K32, (unsigned)N, 32, 32);
  }
  int buf = 0;
  for (int t = 0; t < nk; ++t) {
    if (wv == 0) {
      if (t + 1 < nk) {       // prefetch next K-tile into the other buffer
        int k0 = (t + 1) << 5;
        tdm_load_2d(A + (size_t)blockRow * K32 + k0,
                    (unsigned)(size_t)&As[buf ^ 1][0][0],
                    (unsigned)K32, (unsigned)Mrows, 32, 128);
        tdm_load_2d(W + (size_t)tileN * K32 + k0,
                    (unsigned)(size_t)&Ws[buf ^ 1][0][0],
                    (unsigned)K32, (unsigned)N, 32, 32);
        __builtin_amdgcn_s_wait_tensorcnt(2);  // current pair complete
      } else {
        __builtin_amdgcn_s_wait_tensorcnt(0);
      }
    }
    __syncthreads();
    // fragments per ISA 7.12.2 wave32 layouts
    v16h a0, a1, bf;
#pragma unroll
    for (int e = 0; e < 16; ++e) {
      int ka = ((e >> 3) << 4) + akb + (e & 7);
      a0[e] = As[buf][arow][ka];
      a1[e] = As[buf][arow + 16][ka];
      bf[e] = Ws[buf][(wn << 4) + ncol][ka];
    }
    acc[0] = __builtin_amdgcn_wmma_f32_16x16x32_f16(false, a0, false, bf,
                                                    (short)0, acc[0], false, false);
    acc[1] = __builtin_amdgcn_wmma_f32_16x16x32_f16(false, a1, false, bf,
                                                    (short)0, acc[1], false, false);
    __syncthreads();
    buf ^= 1;
  }

  // epilogue: D layout — VGPR v -> M = v + 8*(lane>=16), N = lane&15
  const int gn   = tileN + (wn << 4) + ncol;
  const float bv = bias ? bias[gn] : 0.f;
  const int mofs = (lane & 16) ? 8 : 0;
#pragma unroll
  for (int h = 0; h < 2; ++h) {
    int rb = blockRow + (wm << 5) + (h << 4) + mofs;
#pragma unroll
    for (int v = 0; v < 8; ++v) {
      int gm = rb + v;
      float x = acc[h][v] + bv;
      if (act == 1)      x = fmaxf(0.f, bng[gn] * (x * BN_INV_C) + bnb[gn]);
      else if (act == 2) x = fmaxf(0.f, x);
      else if (act == 3) x = 0.5f * x * (1.f + erff(x * 0.70710678118f));
      if (resid) x += resid[(size_t)gm * N + gn];
      C[(size_t)gm * N + gn] = x;
    }
  }
}

// ========================= support kernels =================================

// pc (B,3,N) -> xyz (B,N,3)
__global__ void k_transpose_pc(const float* __restrict__ pc, float* __restrict__ xyz,
                               int N) {
  int i = blockIdx.x * blockDim.x + threadIdx.x;   // over B*N
  int b = i / N, n = i % N;
  const float* src = pc + (size_t)b * 3 * N;
  float* dst = xyz + ((size_t)b * N + n) * 3;
  dst[0] = src[n];
  dst[1] = src[N + n];
  dst[2] = src[2 * N + n];
}

// farthest point sampling: one block per batch
__global__ void k_fps(const float* __restrict__ xyz, int* __restrict__ fidx,
                      int N, int M) {
  int b = blockIdx.x;
  const float* P = xyz + (size_t)b * N * 3;
  __shared__ float dist[2048];
  __shared__ float rv[256];
  __shared__ int   ri[256];
  int t = threadIdx.x;
  for (int n = t; n < N; n += 256) dist[n] = 1e10f;
  __syncthreads();
  int last = 0;
  for (int m = 0; m < M; ++m) {
    if (t == 0) fidx[b * M + m] = last;
    float lx = P[last * 3], ly = P[last * 3 + 1], lz = P[last * 3 + 2];
    float bv = -1.f; int bi = 0;
    for (int n = t; n < N; n += 256) {
      float dx = P[n * 3] - lx, dy = P[n * 3 + 1] - ly, dz = P[n * 3 + 2] - lz;
      float dd = fminf(dist[n], dx * dx + dy * dy + dz * dz);
      dist[n] = dd;
      if (dd > bv) { bv = dd; bi = n; }
    }
    rv[t] = bv; ri[t] = bi;
    __syncthreads();
    for (int s = 128; s > 0; s >>= 1) {
      if (t < s) {
        if (rv[t + s] > rv[t] || (rv[t + s] == rv[t] && ri[t + s] < ri[t])) {
          rv[t] = rv[t + s]; ri[t] = ri[t + s];
        }
      }
      __syncthreads();
    }
    last = ri[0];
    __syncthreads();
  }
}

// cxyz[b,m,:] = xyz[b, fidx[b,m], :]
__global__ void k_gather3(const float* __restrict__ xyz, const int* __restrict__ fidx,
                          float* __restrict__ cxyz, int N, int M) {
  int i = blockIdx.x * blockDim.x + threadIdx.x;   // over B*M
  int b = i / M, m = i % M;
  int s = fidx[b * M + m];
  const float* src = xyz + ((size_t)b * N + s) * 3;
  float* dst = cxyz + ((size_t)b * M + m) * 3;
  dst[0] = src[0]; dst[1] = src[1]; dst[2] = src[2];
}

// iterative top-(sel) nearest selection; skip = #nearest to discard (self)
__global__ void k_knn(const float* __restrict__ Q, const float* __restrict__ P,
                      int* __restrict__ out, int Mq, int Np, int sel, int skip) {
  int bm = blockIdx.x;                    // b*Mq + m
  int b = bm / Mq;
  const float* q  = Q + (size_t)bm * 3;
  const float* pp = P + (size_t)b * Np * 3;
  __shared__ float d[2048];
  __shared__ float rv[256];
  __shared__ int   ri[256];
  int t = threadIdx.x;
  float qx = q[0], qy = q[1], qz = q[2];
  for (int n = t; n < Np; n += 256) {
    float dx = pp[n * 3] - qx, dy = pp[n * 3 + 1] - qy, dz = pp[n * 3 + 2] - qz;
    d[n] = dx * dx + dy * dy + dz * dz;
  }
  __syncthreads();
  int total = sel + skip;
  for (int s2 = 0; s2 < total; ++s2) {
    float bv = 3.4e38f; int bi = 0x7fffffff;
    for (int n = t; n < Np; n += 256) {
      float v = d[n];
      if (v < bv || (v == bv && n < bi)) { bv = v; bi = n; }
    }
    rv[t] = bv; ri[t] = bi;
    __syncthreads();
    for (int s3 = 128; s3 > 0; s3 >>= 1) {
      if (t < s3) {
        if (rv[t + s3] < rv[t] || (rv[t + s3] == rv[t] && ri[t + s3] < ri[t])) {
          rv[t] = rv[t + s3]; ri[t] = ri[t + s3];
        }
      }
      __syncthreads();
    }
    int pick = ri[0];
    if (t == 0) {
      if (s2 >= skip) out[(size_t)bm * sel + (s2 - skip)] = pick;
      d[pick] = 3.4e38f;
    }
    __syncthreads();
  }
}

// g[(bm*16+k), ch] = ch<C ? cfeat[ch] : gfeat[ch-C]-cfeat[ch-C]
__global__ void k_build_group(const float* __restrict__ feat, const int* __restrict__ fidx,
                              const int* __restrict__ knn, float* __restrict__ g,
                              int Nin, int M, int Cin, long total) {
  long i = (long)blockIdx.x * blockDim.x + threadIdx.x;
  if (i >= total) return;
  int C2 = 2 * Cin;
  int ch = (int)(i % C2);
  long r = i / C2;
  int k = (int)(r & 15);
  long bm = r >> 4;
  int b = (int)(bm / M), m = (int)(bm % M);
  const float* fb = feat + (size_t)b * Nin * Cin;
  int ci = fidx[b * M + m];
  float val;
  if (ch < Cin) {
    val = fb[(size_t)ci * Cin + ch];
  } else {
    int cc = ch - Cin;
    int ni = knn[bm * 16 + k];
    val = fb[(size_t)ni * Cin + cc] - fb[(size_t)ci * Cin + cc];
  }
  g[i] = val;
}

// max+mean over the 16 grouped rows
__global__ void k_pool(const float* __restrict__ h, float* __restrict__ mx,
                       float* __restrict__ avg, long BM, int C) {
  long i = (long)blockIdx.x * blockDim.x + threadIdx.x;  // over BM*C
  if (i >= BM * (long)C) return;
  int ch = (int)(i % C);
  long bm = i / C;
  const float* base = h + ((size_t)bm * 16) * C + ch;
  float m = -3.4e38f, s = 0.f;
  for (int k = 0; k < 16; ++k) {
    float v = base[(size_t)k * C];
    m = fmaxf(m, v); s += v;
  }
  mx[i] = m;
  avg[i] = s * (1.f / 16.f);
}

// geo features (118 per point): [x(3), knn_xyz(48), rel(48), dist(16), geo(3)]
__global__ void k_geofeat(const float* __restrict__ cxyz, const int* __restrict__ gknn,
                          float* __restrict__ feats, int M) {
  int bm = blockIdx.x * blockDim.x + threadIdx.x;  // over B*M
  int b = bm / M;
  const float* X = cxyz + (size_t)b * M * 3;
  const float* x = cxyz + (size_t)bm * 3;
  float cx = x[0], cy = x[1], cz = x[2];
  float* F = feats + (size_t)bm * 118;
  F[0] = cx; F[1] = cy; F[2] = cz;
  float a00 = 0, a01 = 0, a02 = 0, a11 = 0, a12 = 0, a22 = 0;
  for (int j = 0; j < 16; ++j) {
    int ni = gknn[(size_t)bm * 16 + j];
    float nx = X[ni * 3], ny = X[ni * 3 + 1], nz = X[ni * 3 + 2];
    float rx = nx - cx, ry = ny - cy, rz = nz - cz;
    F[3 + j * 3] = nx; F[4 + j * 3] = ny; F[5 + j * 3] = nz;
    F[51 + j * 3] = rx; F[52 + j * 3] = ry; F[53 + j * 3] = rz;
    F[99 + j] = sqrtf(rx * rx + ry * ry + rz * rz);
    a00 += rx * rx; a01 += rx * ry; a02 += rx * rz;
    a11 += ry * ry; a12 += ry * rz; a22 += rz * rz;
  }
  float inv = 1.f / (15.f + 1e-6f);
  a00 *= inv; a01 *= inv; a02 *= inv; a11 *= inv; a12 *= inv; a22 *= inv;
  float q = (a00 + a11 + a22) / 3.f;
  float p1 = a01 * a01 + a02 * a02 + a12 * a12;
  float p2 = (a00 - q) * (a00 - q) + (a11 - q) * (a11 - q) + (a22 - q) * (a22 - q) + 2.f * p1;
  float p = sqrtf(p2 / 6.f);
  float l1, l2, l3;
  if (p < 1e-12f) {
    l1 = l2 = l3 = q;
  } else {
    float ip = 1.f / p;
    float b00 = (a00 - q) * ip, b11 = (a11 - q) * ip, b22 = (a22 - q) * ip;
    float b01 = a01 * ip, b02 = a02 * ip, b12 = a12 * ip;
    float detB = b00 * (b11 * b22 - b12 * b12) - b01 * (b01 * b22 - b12 * b02)
               + b02 * (b01 * b12 - b11 * b02);
    float r = fminf(1.f, fmaxf(-1.f, 0.5f * detB));
    float phi = acosf(r) / 3.f;
    l1 = q + 2.f * p * cosf(phi);
    l3 = q + 2.f * p * cosf(phi + 2.09439510239f);
    l2 = 3.f * q - l1 - l3;
  }
  float il = 1.f / (l1 + 1e-6f);
  F[115] = (l1 - l2) * il;
  F[116] = (l2 - l3) * il;
  F[117] = l3 * il;
}

// layernorm over last dim; one wave32 per row
__global__ void k_layernorm(const float* __restrict__ x, const float* __restrict__ g,
                            const float* __restrict__ b, float* __restrict__ y, int C) {
  int row = blockIdx.x;
  const float* xr = x + (size_t)row * C;
  int l = threadIdx.x;
  float s = 0.f;
  for (int j = l; j < C; j += 32) s += xr[j];
  for (int o = 16; o > 0; o >>= 1) s += __shfl_xor(s, o);
  float mean = s / (float)C;
  float v = 0.f;
  for (int j = l; j < C; j += 32) { float d = xr[j] - mean; v += d * d; }
  for (int o = 16; o > 0; o >>= 1) v += __shfl_xor(v, o);
  float invs = rsqrtf(v / (float)C + 1e-5f);
  for (int j = l; j < C; j += 32)
    y[(size_t)row * C + j] = (xr[j] - mean) * invs * g[j] + b[j];
}

// softmax attention core, one block per (b,m,head); qkv: (B*M, 3c)
__global__ void k_attn(const float* __restrict__ qkv, float* __restrict__ o,
                       int M, int c, int hd) {
  int id = blockIdx.x;
  int h = id & 3;           // NHEAD = 4
  int bm = id >> 2;
  int b = bm / M;
  int c3 = 3 * c;
  const float* qrow = qkv + (size_t)bm * c3 + h * hd;
  __shared__ float sc[1024];
  __shared__ float red[128];
  int t = threadIdx.x;
  float scale = rsqrtf((float)hd);
  for (int n = t; n < M; n += 128) {
    const float* krow = qkv + ((size_t)b * M + n) * c3 + c + h * hd;
    float a = 0.f;
    for (int d = 0; d < hd; ++d) a += qrow[d] * krow[d];
    sc[n] = a * scale;
  }
  __syncthreads();
  float mv = -3.4e38f;
  for (int n = t; n < M; n += 128) mv = fmaxf(mv, sc[n]);
  red[t] = mv; __syncthreads();
  for (int s = 64; s > 0; s >>= 1) { if (t < s) red[t] = fmaxf(red[t], red[t + s]); __syncthreads(); }
  mv = red[0]; __syncthreads();
  float sum = 0.f;
  for (int n = t; n < M; n += 128) { float e = expf(sc[n] - mv); sc[n] = e; sum += e; }
  red[t] = sum; __syncthreads();
  for (int s = 64; s > 0; s >>= 1) { if (t < s) red[t] += red[t + s]; __syncthreads(); }
  float isum = 1.f / red[0];
  for (int d = t; d < hd; d += 128) {
    float a = 0.f;
    for (int n = 0; n < M; ++n)
      a += sc[n] * qkv[((size_t)b * M + n) * c3 + 2 * c + h * hd + d];
    o[(size_t)bm * c + h * hd + d] = a * isum;
  }
}

// f = concat([mx(c), avg(c), geo(64)])
__global__ void k_concat3(const float* __restrict__ mx, const float* __restrict__ avg,
                          const float* __restrict__ geo, float* __restrict__ f,
                          long BM, int c, int embin) {
  long i = (long)blockIdx.x * blockDim.x + threadIdx.x;
  if (i >= BM * (long)embin) return;
  int ch = (int)(i % embin);
  long bm = i / embin;
  float v;
  if (ch < c)            v = mx[bm * c + ch];
  else if (ch < 2 * c)   v = avg[bm * c + (ch - c)];
  else                   v = geo[bm * 64 + (ch - 2 * c)];
  f[i] = v;
}

// gf[b,ch] = max over M rows
__global__ void k_rowmax(const float* __restrict__ feat, float* __restrict__ gf,
                         int M, int C) {
  int i = blockIdx.x * blockDim.x + threadIdx.x;  // over B*C
  int b = i / C, ch = i % C;
  float m = -3.4e38f;
  for (int r = 0; r < M; ++r) m = fmaxf(m, feat[((size_t)b * M + r) * C + ch]);
  gf[i] = m;
}

// ============================ host side ====================================

struct SP {  // 34 leaves per stage in sorted-key flatten order
  const float *ff1_W, *ff1_b, *ff2_W, *ff2_b, *in_W, *in_b;
  const float *ln12_b, *ln12_g, *ln13_b, *ln13_g, *out_W, *out_b, *proj_W, *proj_b;
  const float *emb_W, *emb_b, *emb_bb, *emb_g;
  const float *geo_W1, *geo_W2, *geo_b1, *geo_b2;
  const float *pa_W0, *pa_W1, *pa_b0, *pa_b1, *pa_bb0, *pa_bb1, *pa_g0, *pa_g1;
  const float *res_W, *res_b, *res_bb, *res_g;
};

static inline void gemm(hipStream_t st, const float* A, const float* W,
                        const float* bias, const float* bng, const float* bnb,
                        const float* resid, float* C, long Mrows, int N, int K, int act,
                        _Float16* Ah, _Float16* Wh) {
  int K32 = (K + 31) & ~31;
  {
    long tot = Mrows * K32;
    k_cvt_pad<<<(unsigned)((tot + 255) / 256), 256, 0, st>>>(A, Ah, K, K32, tot);
  }
  {
    long tot = (long)N * K32;
    k_cvt_pad<<<(unsigned)((tot + 255) / 256), 256, 0, st>>>(W, Wh, K, K32, tot);
  }
  dim3 g((unsigned)(Mrows / 128), (unsigned)(N / 32));
  k_gemm_wmma_f16<<<g, 256, 0, st>>>(Ah, Wh, bias, bng, bnb, resid, C,
                                     (int)Mrows, N, K32, act);
}

extern "C" void kernel_launch(void* const* d_in, const int* in_sizes, int n_in,
                              void* d_out, int out_size, void* d_ws, size_t ws_size,
                              hipStream_t stream) {
  (void)in_sizes; (void)n_in; (void)out_size; (void)ws_size;
  const int B = 16;
  const int CIN[4]  = {3, 64, 128, 256};
  const int CC[4]   = {64, 128, 256, 512};
  const int MSA[4]  = {1024, 512, 256, 128};
  const int EMBIN[4]= {192, 320, 576, 1088};

  const float* pc = (const float*)d_in[0];
  SP sp[4];
  for (int s = 0; s < 4; ++s) {
    const float** f = (const float**)&sp[s];
    for (int j = 0; j < 34; ++j) f[j] = (const float*)d_in[1 + s * 34 + j];
  }

  // ---- workspace carve (assumes d_ws sized generously; ~350 MB used) ----
  char* wsp = (char*)d_ws;
  size_t off = 0;
  auto alloc = [&](size_t nfl) -> float* {
    float* p = (float*)(wsp + off);
    off += ((nfl * 4 + 255) / 256) * 256;
    return p;
  };
  float* xyzIn   = alloc((size_t)B * 2048 * 3);
  float* xyzP0   = alloc((size_t)B * 1024 * 3);
  float* xyzP1   = alloc((size_t)B * 1024 * 3);
  float* featP0  = alloc(1048576);
  float* featP1  = alloc(1048576);
  int*   fidx    = (int*)alloc((size_t)B * 1024);
  int*   knnP    = (int*)alloc((size_t)B * 1024 * 16);
  int*   knnG    = (int*)alloc((size_t)B * 1024 * 16);
  float* bufG    = alloc(33554432);   // grouped features (f32)
  float* bufH    = alloc(16777216);   // MLP hidden / ff hidden (f32)
  _Float16* Ah   = (_Float16*)alloc(16777216);  // 33.5M halves
  _Float16* Wh   = (_Float16*)alloc(524288);    // 1.05M halves
  float* mxB     = alloc(1048576);
  float* mx2B    = alloc(1048576);
  float* avgB    = alloc(1048576);
  float* xproj   = alloc(1048576);
  float* xnB     = alloc(1048576);
  float* x2B     = alloc(1048576);
  float* oB      = alloc(1048576);
  float* qkvB    = alloc(3145728);
  float* geoF    = alloc((size_t)B * 1024 * 118);
  float* geoH1   = alloc(1048576);
  float* geoH2   = alloc(1048576);
  float* catF    = alloc(3145728);
  float* gfB     = alloc((size_t)B * 512);

  // pc (B,3,2048) -> xyz (B,2048,3)
  {
    int tot = B * 2048;
    k_transpose_pc<<<(tot + 255) / 256, 256, 0, stream>>>(pc, xyzIn, 2048);
  }

  const float* curX = xyzIn;
  const float* curF = xyzIn;   // cur_feat = xyz initially
  float* xyzOut[2]  = {xyzP0, xyzP1};
  float* featOut[2] = {featP0, featP1};

  for (int s = 0; s < 4; ++s) {
    const SP& p = sp[s];
    const int Nin = (s == 0) ? 2048 : MSA[s - 1];
    const int Cin = CIN[s];
    const int c   = CC[s];
    const int M   = MSA[s];
    const long BM = (long)B * M;
    float* nxtX = xyzOut[s & 1];
    float* nxtF = featOut[s & 1];

    // ---- patch abstraction ----
    k_fps<<<B, 256, 0, stream>>>(curX, fidx, Nin, M);
    k_gather3<<<(int)((BM + 255) / 256), 256, 0, stream>>>(curX, fidx, nxtX, Nin, M);
    k_knn<<<(int)BM, 256, 0, stream>>>(nxtX, curX, knnP, M, Nin, 16, 0);
    {
      long tot = BM * 16 * (2 * Cin);
      k_build_group<<<(unsigned)((tot + 255) / 256), 256, 0, stream>>>(
          curF, fidx, knnP, bufG, Nin, M, Cin, tot);
    }
    const long rows = BM * 16;
    gemm(stream, bufG, p.pa_W0, p.pa_b0, p.pa_g0, p.pa_bb0, nullptr, bufH,
         rows, c, 2 * Cin, 1, Ah, Wh);
    gemm(stream, bufH, p.pa_W1, p.pa_b1, p.pa_g1, p.pa_bb1, nullptr, bufG,
         rows, c, c, 1, Ah, Wh);
    {
      long tot = BM * c;
      k_pool<<<(unsigned)((tot + 255) / 256), 256, 0, stream>>>(bufG, mxB, avgB, BM, c);
    }
    gemm(stream, mxB, p.res_W, p.res_b, p.res_g, p.res_bb, mxB, mx2B,
         BM, c, c, 1, Ah, Wh);

    // ---- geo encoder on centroids ----
    k_knn<<<(int)BM, 256, 0, stream>>>(nxtX, nxtX, knnG, M, M, 16, 1);
    k_geofeat<<<(int)((BM + 255) / 256), 256, 0, stream>>>(nxtX, knnG, geoF, M);
    gemm(stream, geoF, p.geo_W1, p.geo_b1, nullptr, nullptr, nullptr, geoH1,
         BM, 64, 118, 2, Ah, Wh);
    gemm(stream, geoH1, p.geo_W2, p.geo_b2, nullptr, nullptr, nullptr, geoH2,
         BM, 64, 64, 2, Ah, Wh);

    // ---- sparse attention on avg ----
    gemm(stream, avgB, p.proj_W, p.proj_b, nullptr, nullptr, nullptr, xproj,
         BM, c, c, 0, Ah, Wh);
    k_layernorm<<<(int)BM, 32, 0, stream>>>(xproj, p.ln13_g, p.ln13_b, xnB, c);
    gemm(stream, xnB, p.in_W, p.in_b, nullptr, nullptr, nullptr, qkvB,
         BM, 3 * c, c, 0, Ah, Wh);
    k_attn<<<(int)(BM * 4), 128, 0, stream>>>(qkvB, oB, M, c, c / 4);
    gemm(stream, oB, p.out_W, p.out_b, nullptr, nullptr, xnB, xproj,
         BM, c, c, 0, Ah, Wh);                                   // x = xn + o
    k_layernorm<<<(int)BM, 32, 0, stream>>>(xproj, p.ln12_g, p.ln12_b, x2B, c);
    gemm(stream, x2B, p.ff1_W, p.ff1_b, nullptr, nullptr, nullptr, bufH,
         BM, 1024, c, 3, Ah, Wh);                                // gelu
    gemm(stream, bufH, p.ff2_W, p.ff2_b, nullptr, nullptr, x2B, avgB,
         BM, c, 1024, 0, Ah, Wh);                                // + x2

    // ---- concat + embedding MLP ----
    {
      long tot = BM * EMBIN[s];
      k_concat3<<<(unsigned)((tot + 255) / 256), 256, 0, stream>>>(
          mx2B, avgB, geoH2, catF, BM, c, EMBIN[s]);
    }
    gemm(stream, catF, p.emb_W, p.emb_b, p.emb_g, p.emb_bb, nullptr, nxtF,
         BM, c, EMBIN[s], 1, Ah, Wh);

    curX = nxtX;
    curF = nxtF;
  }

  // final: gf = max over points, then pack (gf, cur_xyz, cur_feat)
  k_rowmax<<<(B * 512 + 255) / 256, 256, 0, stream>>>(curF, gfB, 128, 512);
  float* out = (float*)d_out;
  (void)hipMemcpyAsync(out, gfB, (size_t)B * 512 * sizeof(float),
                       hipMemcpyDeviceToDevice, stream);
  (void)hipMemcpyAsync(out + B * 512, curX, (size_t)B * 128 * 3 * sizeof(float),
                       hipMemcpyDeviceToDevice, stream);
  (void)hipMemcpyAsync(out + B * 512 + B * 128 * 3, curF,
                       (size_t)B * 128 * 512 * sizeof(float),
                       hipMemcpyDeviceToDevice, stream);
}
